// CrossModalMamba_15375982920369
// MI455X (gfx1250) — matrix operand order repaired and checked by reading, not compiled, not run on hardware
//
#include <hip/hip_runtime.h>
#include <hip/hip_bf16.h>

#define PART 6
#define D_STATE 16
#define D_INNER 192
#define DT_RANK 6
#define IN_CDIM 2048
#define BATCH 128
#define HW 288
#define TWO_C 4096
#define NCOLS (BATCH * PART)                // 768 (b,l) columns
#define XPROJ_ROWS (DT_RANK + 2 * D_STATE)  // 38
#define XPROJ_PAD 48                        // padded N for the x_dbl GEMM
#define CATK (96 * PART)                    // 576

typedef __attribute__((ext_vector_type(16))) _Float16 v16h;
typedef __attribute__((ext_vector_type(8)))  _Float16 v8h;
typedef __attribute__((ext_vector_type(8)))  float    v8f;

// ---------------------------------------------------------------------------
// f16 fragment loaders (operands pre-converted in workspace; 16B vector loads).
// A (16x32 MxK): lane half h: elems 0..7  = K = k0+8h .. +7
//                             elems 8..15 = K = k0+8h+16 .. +23   (gap 16)
// B (32x16 KxN): same K layout as A (kbase k0+8h, gap 16)
// ---------------------------------------------------------------------------
__device__ inline v16h load_frag16(const _Float16* __restrict__ p, int kbase, int gap) {
  v8h lo = *(const v8h*)(p + kbase);
  v8h hi = *(const v8h*)(p + kbase + gap);
  return __builtin_shufflevector(lo, hi, 0, 1, 2, 3, 4, 5, 6, 7,
                                 8, 9, 10, 11, 12, 13, 14, 15);
}

// ---------------------------------------------------------------------------
// Kernel 0: one-time f32 -> f16 weight conversion (pads with zeros to n_dst).
// ---------------------------------------------------------------------------
__global__ void cvt_f16_kernel(const float* __restrict__ in, _Float16* __restrict__ out,
                               int n_src, int n_dst) {
  int i = blockIdx.x * blockDim.x + threadIdx.x;
  if (i < n_dst) out[i] = (i < n_src) ? (_Float16)in[i] : (_Float16)0.0f;
}

// ---------------------------------------------------------------------------
// Kernel 1: streaming pooling -> transposed f16 tok  P16[(b*6+l)][4096].
// Thread index: d (=row2) fastest so the 2B f16 stores are fully coalesced;
// each thread reads its 48-float part as 12 aligned float4 (global_load_b128).
// This kernel carries ~604 MB and sets the runtime floor (~26us @ 23.3 TB/s).
// ---------------------------------------------------------------------------
__global__ void pool_kernel(const float* __restrict__ vis,
                            const float* __restrict__ inf,
                            _Float16* __restrict__ poolT16) {
  int pid  = blockIdx.x * blockDim.x + threadIdx.x;   // 0 .. B*6*4096-1
  int row2 = pid % TWO_C;                             // interleaved channel (2c+m)
  int rem  = pid / TWO_C;
  int p    = rem % PART;
  int b    = rem / PART;
  int c    = row2 >> 1;
  const float* src = ((row2 & 1) ? inf : vis) + ((size_t)(b * IN_CDIM + c)) * HW + p * 48;
  const float4* s4 = (const float4*)src;
  float s = 0.0f;
#pragma unroll
  for (int i = 0; i < 12; ++i) {
    float4 v = s4[i];
    s += v.x + v.y + v.z + v.w;
  }
  poolT16[(size_t)(b * PART + p) * TWO_C + row2] = (_Float16)(s * (1.0f / 48.0f));
}

// ---------------------------------------------------------------------------
// Kernel 2: mx = in_proj_w[:192] @ tok (+bias). GEMM 192 x 768 x 4096, f16
// operands / f32 accum; one wave per 16x16 tile, 128 WMMAs (x2 unrolled).
// Writes mx f32 (scan input) and mxT16 (A operand of the x_dbl GEMM).
// ---------------------------------------------------------------------------
__global__ void inproj_gemm_kernel(const _Float16* __restrict__ w16,
                                   const float* __restrict__ bias,
                                   const _Float16* __restrict__ poolT16,
                                   float* __restrict__ mx,
                                   _Float16* __restrict__ mxT16) {
  int wv   = blockIdx.x * (blockDim.x >> 5) + (threadIdx.x >> 5);  // 0..575
  int lane = threadIdx.x & 31;
  int h    = lane >> 4;
  int lm   = lane & 15;
  int tm = wv % 12, tn = wv / 12;
  int m0 = tm * 16, n0 = tn * 16;

  const _Float16* arow = w16 + (size_t)(m0 + lm) * TWO_C;
  int col  = n0 + lm;                                   // (b,l) column
  int bidx = col / PART, ll = col % PART;
  const _Float16* bcol = poolT16 + (size_t)col * TWO_C;

  v8f acc = {};
#pragma unroll 2
  for (int k0 = 0; k0 < TWO_C; k0 += 32) {
    v16h a = load_frag16(arow, k0 + 8 * h, 16);
    v16h bfr = load_frag16(bcol, k0 + 8 * h, 16);
    acc = __builtin_amdgcn_wmma_f32_16x16x32_f16(false, a, false, bfr,
                                                 (short)0, acc, false, false);
  }
#pragma unroll
  for (int r = 0; r < 8; ++r) {
    int row = m0 + 8 * h + r;                           // channel d
    float v = acc[r] + bias[row];
    mx[(size_t)bidx * (D_INNER * PART) + (size_t)row * PART + ll] = v;
    mxT16[(size_t)col * D_INNER + row] = (_Float16)v;
  }
}

// ---------------------------------------------------------------------------
// Kernel 3: x_dbl = xt @ x_proj_w.T. GEMM 768 x 48(pad of 38) x 192.
// K=192 fully unrolled -> 6 static WMMAs; padded xpw16 rows are zero.
// ---------------------------------------------------------------------------
__global__ void xdbl_gemm_kernel(const _Float16* __restrict__ mxT16,
                                 const _Float16* __restrict__ xpw16,
                                 float* __restrict__ xdbl) {
  int wv   = blockIdx.x * (blockDim.x >> 5) + (threadIdx.x >> 5);  // 0..143
  int lane = threadIdx.x & 31;
  int h    = lane >> 4;
  int lm   = lane & 15;
  int tm = wv % 48, tn = wv / 48;
  int m0 = tm * 16, n0 = tn * 16;

  const _Float16* arow = mxT16 + (size_t)(m0 + lm) * D_INNER;  // row = b*6+l
  int j = n0 + lm;                                             // j < 48
  const _Float16* bcol = xpw16 + (size_t)j * D_INNER;

  v8f acc = {};
#pragma unroll
  for (int k0 = 0; k0 < D_INNER; k0 += 32) {
    v16h a = load_frag16(arow, k0 + 8 * h, 16);
    v16h bfr = load_frag16(bcol, k0 + 8 * h, 16);
    acc = __builtin_amdgcn_wmma_f32_16x16x32_f16(false, a, false, bfr,
                                                 (short)0, acc, false, false);
  }
#pragma unroll
  for (int r = 0; r < 8; ++r) {
    int rowOut = m0 + 8 * h + r;                        // = b*6 + l
    if (j < XPROJ_ROWS)
      xdbl[(size_t)rowOut * XPROJ_ROWS + j] = acc[r];
  }
}

// ---------------------------------------------------------------------------
// Kernel 4: selective scan + layernorm, fully fp32. One block per batch,
// thread = channel. u[16] in registers, Bs/Cs broadcast from LDS, layernorm
// via wave shuffles + LDS cross-wave reduction. Emits flat16 (att-GEMM A).
// ---------------------------------------------------------------------------
__global__ void scan_kernel(const float* __restrict__ mx,
                            const float* __restrict__ xdbl,
                            const float* __restrict__ dt_w,
                            const float* __restrict__ A_logs,
                            const float* __restrict__ Ds,
                            const float* __restrict__ ln_g,
                            const float* __restrict__ ln_b,
                            _Float16* __restrict__ flat16) {
  __shared__ float sxd[PART * XPROJ_ROWS];
  __shared__ float rsum[PART][6];
  __shared__ float rsq[PART][6];
  int b = blockIdx.x;
  int d = threadIdx.x;                  // 0..191

  for (int i = d; i < PART * XPROJ_ROWS; i += D_INNER)
    sxd[i] = xdbl[(size_t)b * PART * XPROJ_ROWS + i];
  __syncthreads();

  float dtw[DT_RANK];
#pragma unroll
  for (int r = 0; r < DT_RANK; ++r) dtw[r] = dt_w[d * DT_RANK + r];
  float xrow[PART];
#pragma unroll
  for (int l = 0; l < PART; ++l)
    xrow[l] = mx[(size_t)b * (D_INNER * PART) + (size_t)d * PART + l];
  float As[D_STATE];
#pragma unroll
  for (int n = 0; n < D_STATE; ++n) As[n] = -expf(A_logs[d * D_STATE + n]);

  float dtv[PART];
#pragma unroll
  for (int l = 0; l < PART; ++l) {
    float s = 0.0f;
#pragma unroll
    for (int r = 0; r < DT_RANK; ++r) s += sxd[l * XPROJ_ROWS + r] * dtw[r];
    dtv[l] = (s > 20.0f) ? s : log1pf(expf(s));   // softplus
  }

  float u[D_STATE];
#pragma unroll
  for (int n = 0; n < D_STATE; ++n) u[n] = 0.0f;
  float yv[PART];
#pragma unroll
  for (int l = 0; l < PART; ++l) {
    float e  = dtv[l];
    float ex = e * xrow[l];
    float accy = 0.0f;
#pragma unroll
    for (int n = 0; n < D_STATE; ++n) {
      float dA = expf(e * As[n]);
      u[n] = dA * u[n] + ex * sxd[l * XPROJ_ROWS + DT_RANK + n];
      accy += u[n] * sxd[l * XPROJ_ROWS + DT_RANK + D_STATE + n];
    }
    yv[l] = accy + xrow[l] * Ds[d];
  }

  // Layernorm over the 192 channels for each of the 6 sequence positions.
  int lane = threadIdx.x & 31, wid = threadIdx.x >> 5;
#pragma unroll
  for (int l = 0; l < PART; ++l) {
    float s = yv[l], q = yv[l] * yv[l];
    for (int off = 16; off > 0; off >>= 1) {
      s += __shfl_xor(s, off, 32);
      q += __shfl_xor(q, off, 32);
    }
    if (lane == 0) { rsum[l][wid] = s; rsq[l][wid] = q; }
  }
  __syncthreads();

  float g = ln_g[d], bb = ln_b[d];
  int kd = d >> 1, md = d & 1;          // even channels -> vis, odd -> inf
  _Float16* fo = flat16 + (size_t)md * (BATCH * CATK) + (size_t)b * CATK + kd * PART;
#pragma unroll
  for (int l = 0; l < PART; ++l) {
    float s = 0.0f, q = 0.0f;
#pragma unroll
    for (int w = 0; w < 6; ++w) { s += rsum[l][w]; q += rsq[l][w]; }
    float mu  = s * (1.0f / D_INNER);
    float var = q * (1.0f / D_INNER) - mu * mu;
    fo[l] = (_Float16)((yv[l] - mu) * rsqrtf(var + 1e-5f) * g + bb);
  }
}

// ---------------------------------------------------------------------------
// Kernel 5: attention GEMMs + sigmoid. 2 x (128 x 2048 x 576), f16 operands,
// fp32 accumulate, fused bias + sigmoid epilogue into d_out.
// ---------------------------------------------------------------------------
__global__ void att_gemm_kernel(const _Float16* __restrict__ flat16,
                                const _Float16* __restrict__ catw16,
                                const float* __restrict__ catb,
                                float* __restrict__ out) {
  int wv   = blockIdx.x * (blockDim.x >> 5) + (threadIdx.x >> 5);  // 0..2047
  int lane = threadIdx.x & 31;
  int h    = lane >> 4;
  int lm   = lane & 15;
  int mm = wv >> 10;                   // 0 = vis, 1 = inf
  int t  = wv & 1023;
  int tm = t % 8, tn = t / 8;
  int m0 = tm * 16, n0 = tn * 16;

  const _Float16* arow = flat16 + (size_t)mm * (BATCH * CATK) + (size_t)(m0 + lm) * CATK;
  int o = n0 + lm;
  const _Float16* bcol = catw16 + (size_t)o * CATK;

  v8f acc = {};
#pragma unroll 2
  for (int k0 = 0; k0 < CATK; k0 += 32) {
    v16h a = load_frag16(arow, k0 + 8 * h, 16);
    v16h bfr = load_frag16(bcol, k0 + 8 * h, 16);
    acc = __builtin_amdgcn_wmma_f32_16x16x32_f16(false, a, false, bfr,
                                                 (short)0, acc, false, false);
  }
  float cb = catb[o];
#pragma unroll
  for (int r = 0; r < 8; ++r) {
    int brow = m0 + 8 * h + r;
    float v = acc[r] + cb;
    out[(size_t)mm * (BATCH * IN_CDIM) + (size_t)brow * IN_CDIM + o] =
        1.0f / (1.0f + expf(-v));
  }
}

// ---------------------------------------------------------------------------
extern "C" void kernel_launch(void* const* d_in, const int* in_sizes, int n_in,
                              void* d_out, int out_size, void* d_ws, size_t ws_size,
                              hipStream_t stream) {
  const float* vis       = (const float*)d_in[0];
  const float* inf       = (const float*)d_in[1];
  const float* in_proj_w = (const float*)d_in[2];
  const float* in_proj_b = (const float*)d_in[3];
  const float* x_proj_w  = (const float*)d_in[4];
  const float* dt_w      = (const float*)d_in[5];
  const float* A_logs    = (const float*)d_in[6];
  const float* Ds        = (const float*)d_in[7];
  const float* ln_g      = (const float*)d_in[8];
  const float* ln_b      = (const float*)d_in[9];
  const float* cat_w     = (const float*)d_in[10];
  const float* cat_b     = (const float*)d_in[11];
  float* out = (float*)d_out;

  // Workspace layout (all segment sizes are multiples of 16 bytes).
  char* wsb = (char*)d_ws;
  float* mx         = (float*)wsb;  wsb += (size_t)BATCH * D_INNER * PART * 4;
  float* xdbl       = (float*)wsb;  wsb += (size_t)BATCH * PART * XPROJ_ROWS * 4;
  _Float16* poolT16 = (_Float16*)wsb;  wsb += (size_t)NCOLS * TWO_C * 2;
  _Float16* w16     = (_Float16*)wsb;  wsb += (size_t)D_INNER * TWO_C * 2;
  _Float16* mxT16   = (_Float16*)wsb;  wsb += (size_t)NCOLS * D_INNER * 2;
  _Float16* xpw16   = (_Float16*)wsb;  wsb += (size_t)XPROJ_PAD * D_INNER * 2;
  _Float16* catw16  = (_Float16*)wsb;  wsb += (size_t)IN_CDIM * CATK * 2;
  _Float16* flat16  = (_Float16*)wsb;

  // One-time weight conversions (~4 MB total, negligible vs 604 MB pooling).
  {
    int n = D_INNER * TWO_C;
    cvt_f16_kernel<<<(n + 255) / 256, 256, 0, stream>>>(in_proj_w, w16, n, n);
  }
  {
    int ns = XPROJ_ROWS * D_INNER, nd = XPROJ_PAD * D_INNER;
    cvt_f16_kernel<<<(nd + 255) / 256, 256, 0, stream>>>(x_proj_w, xpw16, ns, nd);
  }
  {
    int n = IN_CDIM * CATK;
    cvt_f16_kernel<<<(n + 255) / 256, 256, 0, stream>>>(cat_w, catw16, n, n);
  }

  pool_kernel<<<(BATCH * TWO_C * PART) / 256, 256, 0, stream>>>(vis, inf, poolT16);
  inproj_gemm_kernel<<<72, 256, 0, stream>>>(w16, in_proj_b, poolT16, mx, mxT16);
  xdbl_gemm_kernel<<<18, 256, 0, stream>>>(mxT16, xpw16, xdbl);
  scan_kernel<<<BATCH, D_INNER, 0, stream>>>(mx, xdbl, dt_w, A_logs, Ds, ln_g, ln_b, flat16);
  att_gemm_kernel<<<256, 256, 0, stream>>>(flat16, catw16, cat_b, out);
}
